// CausalSelfAttention_47734266528028
// MI455X (gfx1250) — compile-verified, matrix operand order checked by reading, not run on hardware
//
#include <hip/hip_runtime.h>

// Causal self-attention forward for MI455X (gfx1250, wave32, WMMA).
// Pipeline: f32->bf16 convert (+weight transpose) -> QKV GEMM (WMMA bf16,
// async-to-LDS double buffered) -> flash attention (WMMA bf16, online softmax,
// async-to-LDS staging) -> output projection (WMMA bf16).

typedef unsigned short u16;
typedef __attribute__((ext_vector_type(16))) __bf16 v16bf;
typedef __attribute__((ext_vector_type(8)))  float  v8f;
typedef __attribute__((ext_vector_type(4)))  int    v4i;

#define BATCH  4
#define SEQ    2048
#define DMODEL 1024
#define NHEAD  16
#define HDIM   64
#define KDIM   1024   // contraction dim of both GEMMs

// ---------- CDNA5 async global->LDS path ----------
#define AS1 __attribute__((address_space(1)))
#define AS3 __attribute__((address_space(3)))

#if defined(__HIP_DEVICE_COMPILE__) && \
    __has_builtin(__builtin_amdgcn_global_load_async_to_lds_b128)
#define HAVE_ASYNC 1
#else
#define HAVE_ASYNC 0
#endif

// Copy 16 bytes global->LDS (per lane). Async variant bypasses VGPRs and is
// tracked with ASYNCcnt so the DMA overlaps WMMA work on the previous tile.
__device__ __forceinline__ void cp16(u16* lds, const u16* g) {
#if HAVE_ASYNC
  __builtin_amdgcn_global_load_async_to_lds_b128(
      (AS1 v4i*)(unsigned long long)g,
      (AS3 v4i*)(unsigned int)(unsigned long long)lds, 0, 0);
#else
  *(uint4*)lds = *(const uint4*)g;
#endif
}

__device__ __forceinline__ void async_wait_all() {
#if HAVE_ASYNC
#if __has_builtin(__builtin_amdgcn_s_wait_asynccnt)
  __builtin_amdgcn_s_wait_asynccnt(0);
#else
  asm volatile("s_wait_asynccnt 0x0" ::: "memory");
#endif
#endif
}

// ---------- helpers ----------
__device__ __forceinline__ u16 f2bf(float f) {
  unsigned int x = __float_as_uint(f);
  unsigned int r = x + 0x7FFFu + ((x >> 16) & 1u);  // round-to-nearest-even
  return (u16)(r >> 16);
}

// Load a 16x32 bf16 WMMA A-fragment (or B^T fragment) from LDS.
// Per ISA 7.12.2: lanes 0-15 row r, K={0..7,16..23}; lanes 16-31 row r, K+=8.
__device__ __forceinline__ v16bf load_frag(const u16* base, int stride) {
  const int lane = threadIdx.x & 31;
  const int r  = lane & 15;
  const int hl = lane >> 4;
  union { v16bf v; unsigned int d[8]; } u;
  const u16* row = base + r * stride;
#pragma unroll
  for (int p = 0; p < 8; ++p) {
    const int kk = ((p & 4) << 2) + hl * 8 + (p & 3) * 2;
    u.d[p] = *(const unsigned int*)(row + kk);
  }
  return u.v;
}

// ---------- conversion kernels ----------
__global__ void f32_to_bf16_kernel(const float* __restrict__ src,
                                   u16* __restrict__ dst, int n) {
  int i = blockIdx.x * blockDim.x + threadIdx.x;
  if (i < n) dst[i] = f2bf(src[i]);
}

// src [rows][cols] f32 -> dst [cols][rows] bf16 (transpose so K is contiguous)
__global__ void transpose_f32_to_bf16_kernel(const float* __restrict__ src,
                                             u16* __restrict__ dst,
                                             int rows, int cols) {
  int i = blockIdx.x * blockDim.x + threadIdx.x;
  if (i < rows * cols) {
    int k = i / cols, n = i - k * cols;
    dst[(size_t)n * rows + k] = f2bf(src[i]);
  }
}

// ---------- GEMM: C = A[M,K] * BT[N,K]^T  (+bias), bf16 in, f32 acc ----------
// Double-buffered LDS, async-to-LDS prefetch of tile k+1 overlapping WMMAs on
// tile k.  EPI==0: QKV scatter epilogue; EPI==1: f32 store (final projection).
template <int EPI>
__global__ void __launch_bounds__(256)
gemm_bf16_wmma(const u16* __restrict__ A, const u16* __restrict__ BT,
               const float* __restrict__ bias,
               u16* __restrict__ qb, u16* __restrict__ kb, u16* __restrict__ vtb,
               float* __restrict__ outf) {
  constexpr int BM = 128, BN = 128, BK = 32, LDT = BK + 8;  // pad: bank-safe
  __shared__ u16 Asb[2][BM * LDT];   // 2 x 10 KB
  __shared__ u16 Bsb[2][BN * LDT];   // 2 x 10 KB

  const int tid  = threadIdx.x;
  const int wid  = tid >> 5;
  const int lane = tid & 31;
  const int wm   = wid >> 1;        // 4 wave rows  -> 32 M each
  const int wn   = wid & 1;         // 2 wave cols  -> 64 N each
  const int m0 = blockIdx.y * BM;
  const int n0 = blockIdx.x * BN;

  auto stage = [&](int kq, int sel) {
#pragma unroll
    for (int i = 0; i < 2; ++i) {
      int c   = tid + i * 256;
      int row = c >> 2;
      int off = (c & 3) * 8;
      cp16(&Asb[sel][row * LDT + off], A + (size_t)(m0 + row) * KDIM + kq + off);
      cp16(&Bsb[sel][row * LDT + off], BT + (size_t)(n0 + row) * KDIM + kq + off);
    }
  };

  v8f acc[2][4] = {};
  constexpr int KT = KDIM / BK;     // 32 k-tiles
  stage(0, 0);

  for (int kt = 0; kt < KT; ++kt) {
    async_wait_all();               // tile kt resident in LDS
    __syncthreads();                // visible to all waves; prev reads done
    if (kt + 1 < KT) stage((kt + 1) * BK, (kt + 1) & 1);  // overlap DMA

    const u16* As = Asb[kt & 1];
    const u16* Bs = Bsb[kt & 1];
    v16bf afr[2], bfr[4];
#pragma unroll
    for (int mi = 0; mi < 2; ++mi)
      afr[mi] = load_frag(As + (wm * 32 + mi * 16) * LDT, LDT);
#pragma unroll
    for (int ni = 0; ni < 4; ++ni)
      bfr[ni] = load_frag(Bs + (wn * 64 + ni * 16) * LDT, LDT);
#pragma unroll
    for (int mi = 0; mi < 2; ++mi)
#pragma unroll
      for (int ni = 0; ni < 4; ++ni)
        acc[mi][ni] = __builtin_amdgcn_wmma_f32_16x16x32_bf16(
            false, afr[mi], false, bfr[ni], (short)0, acc[mi][ni], false, false);
  }

  const int r = lane & 15, hl = lane >> 4;
#pragma unroll
  for (int mi = 0; mi < 2; ++mi)
#pragma unroll
    for (int ni = 0; ni < 4; ++ni)
#pragma unroll
      for (int j = 0; j < 8; ++j) {
        const int m = m0 + wm * 32 + mi * 16 + hl * 8 + j;
        const int n = n0 + wn * 64 + ni * 16 + r;
        const float v = acc[mi][ni][j] + bias[n];
        if constexpr (EPI == 0) {
          const u16 bv  = f2bf(v);
          const int seg = n >> 10;          // 0:q 1:k 2:v
          const int d10 = n & 1023;
          const int h = d10 >> 6, d = d10 & 63;
          const int b = m >> 11, s = m & 2047;
          const size_t bh = (size_t)(b * NHEAD + h);
          if (seg == 0)      qb[(bh * SEQ + s) * HDIM + d] = bv;
          else if (seg == 1) kb[(bh * SEQ + s) * HDIM + d] = bv;
          else               vtb[(bh * HDIM + d) * SEQ + s] = bv;  // V transposed
        } else {
          outf[(size_t)m * DMODEL + n] = v;
        }
      }
}

// ---------- flash attention: per (b,h,128-row q tile), bf16 WMMA ----------
__global__ void __launch_bounds__(256)
flash_attn_wmma(const u16* __restrict__ qb, const u16* __restrict__ kb,
                const u16* __restrict__ vtb, u16* __restrict__ ctx) {
  constexpr int BQ = 128, BKV = 64, LDQ = HDIM + 8;  // 72
  __shared__ u16 Qs[BQ * LDQ];          // 18 KB
  __shared__ u16 Ks[BKV * LDQ];         //  9 KB
  __shared__ u16 VTs[HDIM * LDQ];       //  9 KB
  __shared__ u16 Ps[8 * 16 * LDQ];      // 18 KB, per-wave P relayout

  const int tid = threadIdx.x, wid = tid >> 5, lane = tid & 31;
  const int r = lane & 15, hl = lane >> 4;
  const int q0 = blockIdx.x * BQ;
  const int h = blockIdx.y, b = blockIdx.z;
  const size_t bh = (size_t)(b * NHEAD + h);
  const u16* Qg  = qb  + (bh * SEQ + q0) * HDIM;
  const u16* Kg  = kb  + bh * SEQ * HDIM;
  const u16* VTg = vtb + bh * (size_t)HDIM * SEQ;

  // stage Q tile once: 128x64 bf16 (async; first loop-iteration wait covers it)
#pragma unroll
  for (int i = 0; i < 4; ++i) {
    int c = tid + i * 256;
    int row = c >> 3, off = (c & 7) * 8;
    cp16(Qs + row * LDQ + off, Qg + (size_t)row * HDIM + off);
  }

  v8f acc[4] = {};
  float mrow[8], lrow[8];
#pragma unroll
  for (int j = 0; j < 8; ++j) { mrow[j] = -3.0e38f; lrow[j] = 0.f; }

  for (int kv0 = 0; kv0 < q0 + BQ; kv0 += BKV) {      // causal bound
    // stage K [64,64] and VT [64,64] tiles
#pragma unroll
    for (int i = 0; i < 2; ++i) {
      int c = tid + i * 256;
      int row = c >> 3, off = (c & 7) * 8;
      cp16(Ks + row * LDQ + off,  Kg + (size_t)(kv0 + row) * HDIM + off);
      cp16(VTs + row * LDQ + off, VTg + (size_t)row * SEQ + kv0 + off);
    }
    async_wait_all();
    __syncthreads();

    // S = Q @ K^T : each wave does 16q x 64kv
    v8f sc[4] = {};
#pragma unroll
    for (int kk = 0; kk < HDIM; kk += 32) {
      v16bf qa = load_frag(Qs + (wid * 16) * LDQ + kk, LDQ);
#pragma unroll
      for (int ni = 0; ni < 4; ++ni) {
        v16bf kf = load_frag(Ks + (ni * 16) * LDQ + kk, LDQ);
        sc[ni] = __builtin_amdgcn_wmma_f32_16x16x32_bf16(
            false, qa, false, kf, (short)0, sc[ni], false, false);
      }
    }

    // online softmax over this kv tile (rows live in 16-lane halves)
    float pv[4][8];
#pragma unroll
    for (int j = 0; j < 8; ++j) {
      const int q = q0 + wid * 16 + hl * 8 + j;
      float mx = -3.0e38f;
#pragma unroll
      for (int ni = 0; ni < 4; ++ni) {
        const int kv = kv0 + ni * 16 + r;
        const float t = (kv <= q) ? sc[ni][j] * 0.125f : -3.0e38f; // 1/sqrt(64)
        pv[ni][j] = t;
        mx = fmaxf(mx, t);
      }
#pragma unroll
      for (int o = 1; o < 16; o <<= 1) mx = fmaxf(mx, __shfl_xor(mx, o, 32));
      const float mnew  = fmaxf(mrow[j], mx);
      const float alpha = __expf(mrow[j] - mnew);
      mrow[j] = mnew;
      float sum = 0.f;
#pragma unroll
      for (int ni = 0; ni < 4; ++ni) {
        const float p = __expf(pv[ni][j] - mnew);
        pv[ni][j] = p;
        sum += p;
      }
#pragma unroll
      for (int o = 1; o < 16; o <<= 1) sum += __shfl_xor(sum, o, 32);
      lrow[j] = lrow[j] * alpha + sum;
#pragma unroll
      for (int ni = 0; ni < 4; ++ni) acc[ni][j] *= alpha;
      // relayout P (C-frag -> A-frag) through this wave's private LDS region
#pragma unroll
      for (int ni = 0; ni < 4; ++ni)
        Ps[(wid * 16 + hl * 8 + j) * LDQ + ni * 16 + r] = f2bf(pv[ni][j]);
    }

    // O += P @ V  (same-wave DS ops are in order: no barrier needed for Ps)
#pragma unroll
    for (int kk = 0; kk < BKV; kk += 32) {
      v16bf pa = load_frag(Ps + (wid * 16) * LDQ + kk, LDQ);
#pragma unroll
      for (int ni = 0; ni < 4; ++ni) {
        v16bf vf = load_frag(VTs + (ni * 16) * LDQ + kk, LDQ);
        acc[ni] = __builtin_amdgcn_wmma_f32_16x16x32_bf16(
            false, pa, false, vf, (short)0, acc[ni], false, false);
      }
    }
    __syncthreads();
  }

  // ctx[b,s,h*64+d] bf16, row-major -> A operand of the output projection
#pragma unroll
  for (int ni = 0; ni < 4; ++ni)
#pragma unroll
    for (int j = 0; j < 8; ++j) {
      const int s = q0 + wid * 16 + hl * 8 + j;
      const int d = ni * 16 + r;
      ctx[((size_t)b * SEQ + s) * DMODEL + h * HDIM + d] = f2bf(acc[ni][j] / lrow[j]);
    }
}

// ---------- host ----------
extern "C" void kernel_launch(void* const* d_in, const int* in_sizes, int n_in,
                              void* d_out, int out_size, void* d_ws, size_t ws_size,
                              hipStream_t stream) {
  const float* inp    = (const float*)d_in[0];
  const float* w_attn = (const float*)d_in[1];
  const float* b_attn = (const float*)d_in[2];
  const float* w_proj = (const float*)d_in[3];
  const float* b_proj = (const float*)d_in[4];
  float* out = (float*)d_out;

  // workspace partition (bf16 everywhere): ~88 MiB total
  char* ws = (char*)d_ws;
  const size_t MiB = (size_t)1 << 20;
  u16* Xbf    = (u16*)(ws + 0);          // [8192,1024]   16 MiB
  u16* WqkvT  = (u16*)(ws + 16 * MiB);   // [3072,1024]    6 MiB
  u16* WprojT = (u16*)(ws + 22 * MiB);   // [1024,1024]    2 MiB
  u16* Qb     = (u16*)(ws + 24 * MiB);   // [B,H,S,64]    16 MiB
  u16* Kb     = (u16*)(ws + 40 * MiB);   // [B,H,S,64]    16 MiB
  u16* VTb    = (u16*)(ws + 56 * MiB);   // [B,H,64,S]    16 MiB
  u16* CTX    = (u16*)(ws + 72 * MiB);   // [8192,1024]   16 MiB

  const int nX = BATCH * SEQ * DMODEL;
  f32_to_bf16_kernel<<<(nX + 255) / 256, 256, 0, stream>>>(inp, Xbf, nX);
  transpose_f32_to_bf16_kernel<<<(DMODEL * 3 * DMODEL + 255) / 256, 256, 0, stream>>>(
      w_attn, WqkvT, DMODEL, 3 * DMODEL);
  transpose_f32_to_bf16_kernel<<<(DMODEL * DMODEL + 255) / 256, 256, 0, stream>>>(
      w_proj, WprojT, DMODEL, DMODEL);

  gemm_bf16_wmma<0><<<dim3(3 * DMODEL / 128, (BATCH * SEQ) / 128), 256, 0, stream>>>(
      Xbf, WqkvT, b_attn, Qb, Kb, VTb, (float*)nullptr);

  flash_attn_wmma<<<dim3(SEQ / 128, NHEAD, BATCH), 256, 0, stream>>>(Qb, Kb, VTb, CTX);

  gemm_bf16_wmma<1><<<dim3(DMODEL / 128, (BATCH * SEQ) / 128), 256, 0, stream>>>(
      CTX, WprojT, b_proj, (u16*)nullptr, (u16*)nullptr, (u16*)nullptr, out);
}
